// MixtureOfExperts_3229815407333
// MI455X (gfx1250) — matrix operand order repaired and checked by reading, not compiled, not run on hardware
//
#include <hip/hip_runtime.h>
#include <hip/hip_bf16.h>
#include <math.h>

typedef __attribute__((ext_vector_type(16))) _Float16 v16h;
typedef __attribute__((ext_vector_type(8)))  _Float16 v8h;
typedef __attribute__((ext_vector_type(4)))  _Float16 v4h;
typedef __attribute__((ext_vector_type(8)))  float    v8f;

#define D_MODEL  1024
#define FF_DIM   4096
#define NEXP     8
#define TOKENS   4096
#define M_TILE   32
#define FF_CHUNK 512
#define XPITCH   (D_MODEL + 8)    // f16 elems; 2064B rows: 16B aligned, bank-spread
#define HPITCH   (FF_CHUNK + 8)   // f16 elems; 1040B rows: 16B aligned, bank-spread

// Branch-free GELU (tanh form). Max abs deviation from exact erf-GELU ~3e-4,
// below the f16-quantization error already incurred by the WMMA inputs.
__device__ __forceinline__ float gelu_fast(float v) {
    float u = 0.7978845608028654f * (v + 0.044715f * v * v * v);
    float t = 1.0f - 2.0f / (1.0f + __expf(2.0f * u));   // tanh(u), one v_exp_f32
    return 0.5f * v * (1.0f + t);
}

// A-fragment (16x32 f16) from a row-major f16 LDS row.
// Layout: v16h elems 0..15 = K ka..ka+15 (matches B frags),
// ka = kb + (lane>=16 ? 16 : 0); row M = lane&15 (+16 for second m-tile).
__device__ __forceinline__ v16h a_frag_from_lds(const _Float16* p) {
    union { v16h v; v8h h[2]; } u;
    u.h[0] = *(const v8h*)p;
    u.h[1] = *(const v8h*)(p + 8);
    return u.v;
}

// ---------------------------------------------------------------------------
// Router: 1 wave per token. comb[t][e] = softmax-of-top2-probs weights.
// ---------------------------------------------------------------------------
__global__ __launch_bounds__(256) void router_kernel(
    const float* __restrict__ x, const float* __restrict__ Wr,
    const float* __restrict__ br, float* __restrict__ comb)
{
    const int wave = threadIdx.x >> 5;
    const int lane = threadIdx.x & 31;
    const int t    = blockIdx.x * 8 + wave;

    float a[NEXP];
    #pragma unroll
    for (int e = 0; e < NEXP; ++e) a[e] = 0.f;

    const float* xr = x + (size_t)t * D_MODEL;
    for (int d = lane; d < D_MODEL; d += 32) {
        float xv = xr[d];
        const float* wr = Wr + d * NEXP;
        #pragma unroll
        for (int e = 0; e < NEXP; ++e) a[e] += xv * wr[e];
    }
    #pragma unroll
    for (int off = 16; off >= 1; off >>= 1)
        #pragma unroll
        for (int e = 0; e < NEXP; ++e) a[e] += __shfl_xor(a[e], off, 32);

    if (lane == 0) {
        float lg[NEXP], p[NEXP];
        float mx = -1e30f;
        #pragma unroll
        for (int e = 0; e < NEXP; ++e) { lg[e] = a[e] + br[e]; mx = fmaxf(mx, lg[e]); }
        float s = 0.f;
        #pragma unroll
        for (int e = 0; e < NEXP; ++e) { p[e] = expf(lg[e] - mx); s += p[e]; }
        #pragma unroll
        for (int e = 0; e < NEXP; ++e) p[e] /= s;
        int i1 = 0;
        #pragma unroll
        for (int e = 1; e < NEXP; ++e) if (p[e] > p[i1]) i1 = e;
        int i2 = (i1 == 0) ? 1 : 0;
        #pragma unroll
        for (int e = 0; e < NEXP; ++e) if (e != i1 && p[e] > p[i2]) i2 = e;
        // reference renormalizes with softmax over the two top *probabilities*
        float e1 = expf(p[i1]), e2 = expf(p[i2]);
        float inv = 1.f / (e1 + e2);
        float w1 = e1 * inv, w2 = e2 * inv;
        #pragma unroll
        for (int e = 0; e < NEXP; ++e)
            comb[(size_t)t * NEXP + e] = (e == i1) ? w1 : ((e == i2) ? w2 : 0.f);
    }
}

// ---------------------------------------------------------------------------
// Tiled transpose + f32->f16 cast:  in [R,C] row-major  ->  out [C,R]
// grid (C/32, R/32, E), block (32,8)
// ---------------------------------------------------------------------------
__global__ __launch_bounds__(256) void transpose_cast_kernel(
    const float* __restrict__ in, _Float16* __restrict__ out, int R, int C)
{
    __shared__ _Float16 tile[32][33];
    const size_t eoff = (size_t)blockIdx.z * (size_t)R * (size_t)C;
    const float* src = in + eoff;
    _Float16* dst = out + eoff;
    const int c0 = blockIdx.x * 32, r0 = blockIdx.y * 32;
    #pragma unroll
    for (int i = threadIdx.y; i < 32; i += 8)
        tile[i][threadIdx.x] = (_Float16)src[(size_t)(r0 + i) * C + c0 + threadIdx.x];
    __syncthreads();
    #pragma unroll
    for (int i = threadIdx.y; i < 32; i += 8)
        dst[(size_t)(c0 + i) * R + r0 + threadIdx.x] = tile[threadIdx.x][i];
}

// ---------------------------------------------------------------------------
// Fused MoE: per workgroup = 32 tokens; wave w owns output cols [128w,128w+128).
// x tile cached once in LDS as f16; per expert/FF-chunk: GEMM1 -> bias ->
// gelu -> *comb -> LDS H; GEMM2 accumulates into persistent WMMA C tiles.
// ---------------------------------------------------------------------------
__global__ __launch_bounds__(256) void moe_fused_kernel(
    const float* __restrict__ x,   const _Float16* __restrict__ W1T,  // [E][FF][D]
    const float* __restrict__ b1,  const _Float16* __restrict__ W2T,  // [E][D][FF]
    const float* __restrict__ b2,  const float* __restrict__ comb,
    float* __restrict__ out)
{
    __shared__ _Float16 lds_x[M_TILE * XPITCH];   // 66,048 B
    __shared__ _Float16 lds_h[M_TILE * HPITCH];   // 33,280 B
    __shared__ float    lds_comb[M_TILE * NEXP];  //  1,024 B

    const int tid  = threadIdx.x;
    const int wave = tid >> 5;
    const int lane = tid & 31;
    const int l16  = lane & 15;
    const int khw  = lane >> 4;        // 0/1: K-half selector for A/B frags
    const int hi8  = khw * 8;          // C-tile row offset for upper lanes
    const int t0   = blockIdx.x * M_TILE;

    lds_comb[tid] = comb[(size_t)t0 * NEXP + tid];   // 256 == 32*8

    // stage x tile once: f32 global -> f16 LDS (vectorized float4 -> v4h)
    for (int v = tid; v < M_TILE * (D_MODEL / 4); v += 256) {
        const int row = v >> 8;               // D_MODEL/4 == 256
        const int c4  = (v & 255) * 4;
        float4 f = *(const float4*)(x + (size_t)(t0 + row) * D_MODEL + c4);
        v4h h;
        h[0] = (_Float16)f.x; h[1] = (_Float16)f.y;
        h[2] = (_Float16)f.z; h[3] = (_Float16)f.w;
        *(v4h*)(lds_x + row * XPITCH + c4) = h;
    }
    __syncthreads();

    // persistent accumulators, seeded with sum_e comb[t,e]*b2[e,col]
    v8f acc[2][8];
    const int ncol0 = wave * 128;
    #pragma unroll
    for (int mt = 0; mt < 2; ++mt)
        #pragma unroll
        for (int j = 0; j < 8; ++j) {
            const int col = ncol0 + j * 16 + l16;
            #pragma unroll
            for (int r = 0; r < 8; ++r) {
                const int row = mt * 16 + hi8 + r;
                float s = 0.f;
                #pragma unroll
                for (int e = 0; e < NEXP; ++e)
                    s += lds_comb[row * NEXP + e] * b2[e * D_MODEL + col];
                acc[mt][j][r] = s;
            }
        }

    for (int e = 0; e < NEXP; ++e) {
        const _Float16* w1e = W1T + (size_t)e * FF_DIM * D_MODEL;
        const _Float16* w2e = W2T + (size_t)e * D_MODEL * FF_DIM;
        const float*    b1e = b1 + e * FF_DIM;

        for (int c = 0; c < FF_DIM / FF_CHUNK; ++c) {
            const int ffb = c * FF_CHUNK;

            // ---- GEMM1: this wave computes H[:, wave*64 .. +64) of the chunk
            v8f c1[2][4];
            #pragma unroll
            for (int mt = 0; mt < 2; ++mt)
                #pragma unroll
                for (int nt = 0; nt < 4; ++nt)
                    #pragma unroll
                    for (int r = 0; r < 8; ++r) c1[mt][nt][r] = 0.f;

            for (int kb = 0; kb < D_MODEL; kb += 32) {
                v16h a0 = a_frag_from_lds(lds_x + (0  + l16) * XPITCH + kb + khw * 16);
                v16h a1 = a_frag_from_lds(lds_x + (16 + l16) * XPITCH + kb + khw * 16);
                #pragma unroll
                for (int nt = 0; nt < 4; ++nt) {
                    const _Float16* bp = w1e
                        + (size_t)(ffb + wave * 64 + nt * 16 + l16) * D_MODEL
                        + kb + khw * 16;
                    v16h bf = *(const v16h*)bp;                       // 32B contiguous
                    c1[0][nt] = __builtin_amdgcn_wmma_f32_16x16x32_f16(
                        false, a0, false, bf, (short)0, c1[0][nt], false, false);
                    c1[1][nt] = __builtin_amdgcn_wmma_f32_16x16x32_f16(
                        false, a1, false, bf, (short)0, c1[1][nt], false, false);
                }
            }

            // bias + fast GELU + combine-weight scale -> LDS (f16), branch-free
            #pragma unroll
            for (int nt = 0; nt < 4; ++nt) {
                const int colc = wave * 64 + nt * 16 + l16;
                const float b1v = b1e[ffb + colc];
                #pragma unroll
                for (int mt = 0; mt < 2; ++mt)
                    #pragma unroll
                    for (int r = 0; r < 8; ++r) {
                        const int row = mt * 16 + hi8 + r;
                        float v = c1[mt][nt][r] + b1v;
                        lds_h[row * HPITCH + colc] =
                            (_Float16)(lds_comb[row * NEXP + e] * gelu_fast(v));
                    }
            }
            __syncthreads();

            // ---- GEMM2 partial: acc += H_chunk @ W2[ffb:+512, ncol0:+128]
            for (int kk = 0; kk < FF_CHUNK; kk += 32) {
                v16h a0 = a_frag_from_lds(lds_h + (0  + l16) * HPITCH + kk + khw * 16);
                v16h a1 = a_frag_from_lds(lds_h + (16 + l16) * HPITCH + kk + khw * 16);
                #pragma unroll
                for (int j = 0; j < 8; ++j) {
                    const _Float16* bp = w2e
                        + (size_t)(ncol0 + j * 16 + l16) * FF_DIM
                        + ffb + kk + khw * 16;
                    v16h bf = *(const v16h*)bp;                       // 32B contiguous
                    acc[0][j] = __builtin_amdgcn_wmma_f32_16x16x32_f16(
                        false, a0, false, bf, (short)0, acc[0][j], false, false);
                    acc[1][j] = __builtin_amdgcn_wmma_f32_16x16x32_f16(
                        false, a1, false, bf, (short)0, acc[1][j], false, false);
                }
            }
            __syncthreads();   // lds_h reused next chunk
        }
    }

    // write final f32 output
    #pragma unroll
    for (int mt = 0; mt < 2; ++mt)
        #pragma unroll
        for (int j = 0; j < 8; ++j) {
            const int col = ncol0 + j * 16 + l16;
            #pragma unroll
            for (int r = 0; r < 8; ++r) {
                const int row = mt * 16 + hi8 + r;
                out[(size_t)(t0 + row) * D_MODEL + col] = acc[mt][j][r];
            }
        }
}

// ---------------------------------------------------------------------------
extern "C" void kernel_launch(void* const* d_in, const int* in_sizes, int n_in,
                              void* d_out, int out_size, void* d_ws, size_t ws_size,
                              hipStream_t stream) {
    (void)in_sizes; (void)n_in; (void)out_size; (void)ws_size;
    const float* x  = (const float*)d_in[0];
    const float* W1 = (const float*)d_in[1];
    const float* b1 = (const float*)d_in[2];
    const float* W2 = (const float*)d_in[3];
    const float* b2 = (const float*)d_in[4];
    const float* Wr = (const float*)d_in[5];
    const float* br = (const float*)d_in[6];
    float* out = (float*)d_out;

    // workspace layout: comb (128KB) | W1T f16 (64MB) | W2T f16 (64MB)
    char* ws = (char*)d_ws;
    float*    comb = (float*)ws;
    _Float16* W1T  = (_Float16*)(ws + (size_t)(1 << 20));
    _Float16* W2T  = (_Float16*)(ws + (size_t)(1 << 20)
                                    + (size_t)NEXP * FF_DIM * D_MODEL * sizeof(_Float16));

    router_kernel<<<TOKENS / 8, 256, 0, stream>>>(x, Wr, br, comb);
    // W1 [E][D][FF] -> W1T [E][FF][D]
    transpose_cast_kernel<<<dim3(FF_DIM / 32, D_MODEL / 32, NEXP), dim3(32, 8), 0, stream>>>(
        W1, W1T, D_MODEL, FF_DIM);
    // W2 [E][FF][D] -> W2T [E][D][FF]
    transpose_cast_kernel<<<dim3(D_MODEL / 32, FF_DIM / 32, NEXP), dim3(32, 8), 0, stream>>>(
        W2, W2T, FF_DIM, D_MODEL);
    moe_fused_kernel<<<TOKENS / M_TILE, 256, 0, stream>>>(x, W1T, b1, W2T, b2, comb, out);
}
